// Layer_VGIB_86878598464008
// MI455X (gfx1250) — matrix operand order repaired and checked by reading, not compiled, not run on hardware
//
#include <hip/hip_runtime.h>
#include <hip/hip_bf16.h>
#include <math.h>

#define B_    64
#define NAP   256
#define NUE   128
#define DEG   32
#define EAP   (NAP*DEG)
#define EUE   (NUE*DEG)
#define D1    64
#define D2    32
#define OUT_  64
#define CH    (2*OUT_)
#define S_    10

typedef _Float16 v16h __attribute__((ext_vector_type(16)));
typedef _Float16 v8h  __attribute__((ext_vector_type(8)));
typedef float    v8f  __attribute__((ext_vector_type(8)));

// ---------------- deterministic hash RNG ----------------
__device__ __forceinline__ unsigned pcg(unsigned x) {
    x = x * 747796405u + 2891336453u;
    unsigned w = ((x >> ((x >> 28u) + 4u)) ^ x) * 277803737u;
    return (w >> 22u) ^ w;
}
__device__ __forceinline__ float u01(unsigned h) {
    return ((float)h + 0.5f) * 2.3283064365386963e-10f; // (0,1)
}
__device__ __forceinline__ float hash_normal(unsigned key) {
    float u1 = u01(pcg(key));
    float u2 = u01(pcg(key ^ 0x68bc21ebu));
    return sqrtf(-2.0f * __logf(u1)) * __cosf(6.28318530718f * u2);
}

// ---------------- 1) per-node attention scores ----------------
__global__ void node_score_kernel(const float* __restrict__ A,
                                  const float* __restrict__ att,
                                  float* __restrict__ s, int N) {
    int idx = blockIdx.x * blockDim.x + threadIdx.x;
    if (idx >= B_ * N) return;
    int b = idx / N, n = idx - b * N;
    const float* Ab = A + (size_t)b * D1 * N;
    float si = 0.f, sj = 0.f;
    #pragma unroll 8
    for (int d = 0; d < D1; ++d) {
        float v = Ab[(size_t)d * N + n];
        si += v * att[d];
        sj += v * att[D1 + d];
    }
    s[2 * idx]     = si;
    s[2 * idx + 1] = sj;
}

// ---------------- 2) per-edge gate + alpha ----------------
__global__ void edge_kernel(const float* __restrict__ s,
                            const int* __restrict__ graph,
                            float* __restrict__ gate,
                            float* __restrict__ alpha_out,
                            int N, int E, unsigned seed) {
    int idx = blockIdx.x * blockDim.x + threadIdx.x;
    if (idx >= B_ * E) return;
    int b = idx / E, e = idx - b * E;
    int src = graph[(size_t)b * 2 * E + e];
    int dst = graph[(size_t)b * 2 * E + E + e];
    float score = s[2 * (b * N + dst)] + s[2 * (b * N + src) + 1];
    float lr = score > 0.f ? score : 0.2f * score;          // leaky_relu(0.2)
    float a  = 1.f / (1.f + __expf(-lr));
    a = fminf(fmaxf(a, 0.01f), 0.99f);
    float logits = __logf(a) - __logf(1.f - a);
    float u = u01(pcg((unsigned)idx * 2654435761u + seed));
    float noise = __logf(u) - __logf(1.f - u);              // logistic sample
    gate[idx]      = 1.f / (1.f + __expf(-(logits + noise) * 10.f));
    alpha_out[idx] = a;
}

// ---------------- 3) KL(alpha || 0.5) summed over edges, per batch ----------------
__global__ void kl_kernel(const float* __restrict__ alpha, float* __restrict__ out, int E) {
    int b = blockIdx.x, t = threadIdx.x;
    const float LOG_HALF = -0.6931471805599453f;
    float acc = 0.f;
    for (int e = t; e < E; e += 256) {
        float a = alpha[(size_t)b * E + e];
        acc += a * (__logf(a) - LOG_HALF) + (1.f - a) * (__logf(1.f - a) - LOG_HALF);
    }
    __shared__ float red[256];
    red[t] = acc; __syncthreads();
    for (int st = 128; st > 0; st >>= 1) { if (t < st) red[t] += red[t + st]; __syncthreads(); }
    if (t == 0) out[b] = red[0];
}

// ---------------- 4) edge aggregation -> transposed f16 features ----------------
__global__ __launch_bounds__(64) void agg_kernel(const float* __restrict__ A,
                                                 const float* __restrict__ gate,
                                                 const int* __restrict__ graph,
                                                 const int* __restrict__ gfa,
                                                 _Float16* __restrict__ AT,
                                                 int N, int E) {
    int bn = blockIdx.x;
    int b = bn / N, n = bn - b * N;
    __shared__ float gsh[DEG];
    __shared__ int   ssh[DEG];
    int t = threadIdx.x;
    if (t < DEG) {
        int e  = gfa[((size_t)b * N + n) * DEG + t];
        gsh[t] = gate[(size_t)b * E + e];
        ssh[t] = graph[(size_t)b * 2 * E + e];   // row 0 = source
    }
    __syncthreads();
    const float* Ab = A + (size_t)b * D1 * N;
    float acc = 0.f;
    #pragma unroll 8
    for (int k = 0; k < DEG; ++k) acc += gsh[k] * Ab[(size_t)t * N + ssh[k]];
    AT[((size_t)b * N + n) * D1 + t] = (_Float16)acc;
}

// ---------------- 5) feature column means (over nodes) ----------------
__global__ void colmean_kernel(const _Float16* __restrict__ AT, float* __restrict__ mean, int N) {
    int idx = blockIdx.x * blockDim.x + threadIdx.x;
    if (idx >= B_ * D1) return;
    int b = idx / D1, d = idx - b * D1;
    float acc = 0.f;
    for (int n = 0; n < N; ++n) acc += (float)AT[((size_t)b * N + n) * D1 + d];
    mean[idx] = acc / (float)N;
}

// ---------------- 6) H means along both trailing axes (single L2-friendly pass) ----------------
__global__ __launch_bounds__(256) void hmean_kernel(const float* __restrict__ H,
                                                    _Float16* __restrict__ HmApT,
                                                    _Float16* __restrict__ HmUeT) {
    int bd = blockIdx.x;                       // b*D2 + d2
    int b = bd / D2, d2 = bd - b * D2;
    const float* Hs = H + (size_t)bd * NAP * NUE;
    int t = threadIdx.x;
    // row mean over NUE (contiguous), one AP node per thread
    {
        const float* row = Hs + (size_t)t * NUE;
        float acc = 0.f;
        for (int c = 0; c < NUE; c += 4) {
            __builtin_prefetch(row + c + 64, 0, 1);
            acc += row[c] + row[c + 1] + row[c + 2] + row[c + 3];
        }
        HmApT[((size_t)b * NAP + t) * D2 + d2] = (_Float16)(acc * (1.f / NUE));
    }
    // column mean over NAP (strided, coalesced across lanes); slab is L2-hot now
    __shared__ float cs[NUE];
    int c = t & 127, half = t >> 7;
    float acc2 = 0.f;
    for (int n = half * 128; n < half * 128 + 128; ++n) acc2 += Hs[(size_t)n * NUE + c];
    if (half == 0) cs[c] = acc2;
    __syncthreads();
    if (half == 1)
        HmUeT[((size_t)b * NUE + c) * D2 + d2] = (_Float16)((cs[c] + acc2) * (1.f / NAP));
}

// ---------------- 7) weight conversion with scale folding ----------------
__global__ void convw_kernel(const float* __restrict__ Q1, const float* __restrict__ P1,
                             _Float16* __restrict__ Qc, _Float16* __restrict__ Pc) {
    int t = blockIdx.x * blockDim.x + threadIdx.x;
    if (t < CH * D1) Qc[t] = (_Float16)(2.0f * Q1[t]);
    if (t < CH * D2) Pc[t] = (_Float16)(0.1f * P1[t]);
}

// ---------------- 8) cross-branch bias: 2 * Q2 @ otherMean ----------------
__global__ void bias_kernel(const float* __restrict__ Q2, const float* __restrict__ mean,
                            float* __restrict__ bias) {
    int idx = blockIdx.x * blockDim.x + threadIdx.x;
    if (idx >= B_ * CH) return;
    int b = idx / CH, o = idx - b * CH;
    float acc = 0.f;
    #pragma unroll 8
    for (int d = 0; d < D1; ++d) acc += Q2[o * D1 + d] * mean[b * D1 + d];
    bias[idx] = 2.0f * acc;
}

// ---------------- 9) WMMA GEMM: out = ReLU(Qc@Aap + Pc@Hm + bias) ----------------
// Each wave: one 16-row M tile x FOUR 16-col N tiles (A-frag reuse, 12 wmma).
// Interleaved accumulators -> no WMMA->WMMA RAW hazard between consecutive ops.
// N is a template param so epilogue stores use immediate offsets.
template <int N>
__global__ __launch_bounds__(256) void gemm_wmma_kernel(
    const _Float16* __restrict__ Qc,   // CH x D1 row-major f16 (x2 folded)
    const _Float16* __restrict__ Pc,   // CH x D2 row-major f16 (x0.1 folded)
    const _Float16* __restrict__ BT,   // B x N x D1  (Aap transposed, f16)
    const _Float16* __restrict__ HT,   // B x N x D2  (H-mean transposed, f16)
    const float*    __restrict__ bias, // B x CH
    float*          __restrict__ outbuf) { // B x CH x N (post-ReLU, pre-BN)
    constexpr int NT4 = N / 64;                    // groups of 4 N-tiles
    constexpr int wavesPerBatch = (CH / 16) * NT4;
    int wave = blockIdx.x * (blockDim.x >> 5) + (threadIdx.x >> 5);
    int b = wave / wavesPerBatch;
    if (b >= B_) return;
    int tile = wave - b * wavesPerBatch;
    int mt  = tile / NT4;
    int nt4 = tile - mt * NT4;
    int lane = threadIdx.x & 31;
    int mrow = mt * 16 + (lane & 15);
    int nbase = nt4 * 64 + (lane & 15);
    int k0 = (lane >> 4) * 16;   // A-frag K base (contiguous, matches B)
    int kb = (lane >> 4) * 16;   // B-frag K base     (contiguous, matches A)

    // A fragments (reused across the 4 N tiles)
    const _Float16* qrow = Qc + (size_t)mrow * D1;
    const _Float16* prow = Pc + (size_t)mrow * D2;
    v8h q00 = *(const v8h*)(qrow + k0);
    v8h q01 = *(const v8h*)(qrow + 8 + k0);
    v8h q10 = *(const v8h*)(qrow + 32 + k0);
    v8h q11 = *(const v8h*)(qrow + 40 + k0);
    v8h p0  = *(const v8h*)(prow + k0);
    v8h p1  = *(const v8h*)(prow + 8 + k0);
    v16h afrag[3];
    afrag[0] = __builtin_shufflevector(p0,  p1,  0,1,2,3,4,5,6,7,8,9,10,11,12,13,14,15);
    afrag[1] = __builtin_shufflevector(q00, q01, 0,1,2,3,4,5,6,7,8,9,10,11,12,13,14,15);
    afrag[2] = __builtin_shufflevector(q10, q11, 0,1,2,3,4,5,6,7,8,9,10,11,12,13,14,15);

    // B fragments for 4 tiles x 3 K-steps
    v16h bfrag[3][4];
    #pragma unroll
    for (int j = 0; j < 4; ++j) {
        int ncol = nbase + j * 16;
        const _Float16* brow = BT + ((size_t)b * N + ncol) * D1;
        const _Float16* hrow = HT + ((size_t)b * N + ncol) * D2;
        bfrag[0][j] = *(const v16h*)(hrow + kb);        // Hm, K=0..31
        bfrag[1][j] = *(const v16h*)(brow + kb);        // Aap, K=0..31
        bfrag[2][j] = *(const v16h*)(brow + 32 + kb);   // Aap, K=32..63
    }

    v8f c[4] = {v8f{}, v8f{}, v8f{}, v8f{}};
    #pragma unroll
    for (int s = 0; s < 3; ++s)
        #pragma unroll
        for (int j = 0; j < 4; ++j)
            c[j] = __builtin_amdgcn_wmma_f32_16x16x32_f16(
                false, afrag[s], false, bfrag[s][j], (short)0, c[j], false, false);

    int mbase = mt * 16 + ((lane >> 4) ? 8 : 0);
    v8f bv = *(const v8f*)(bias + b * CH + mbase);       // 8 contiguous channels
    #pragma unroll
    for (int j = 0; j < 4; ++j) {
        int ncol = nbase + j * 16;
        float* outp = outbuf + ((size_t)b * CH + mbase) * N + ncol;
        #pragma unroll
        for (int r = 0; r < 8; ++r)
            outp[(size_t)r * N] = fmaxf(c[j][r] + bv[r], 0.f);   // immediate offsets
    }
}

// ---------------- 10) BN stats per channel over (b, n) ----------------
__global__ void bn_stats_kernel(const float* __restrict__ buf, float* __restrict__ stats, int N) {
    int o = blockIdx.x, t = threadIdx.x;
    int tot = B_ * N;
    float s = 0.f, s2 = 0.f;
    for (int i = t; i < tot; i += 256) {
        int b = i / N, n = i - b * N;
        float v = buf[((size_t)b * CH + o) * N + n];
        s += v; s2 += v * v;
    }
    __shared__ float r1[256], r2[256];
    r1[t] = s; r2[t] = s2; __syncthreads();
    for (int st = 128; st > 0; st >>= 1) {
        if (t < st) { r1[t] += r1[t + st]; r2[t] += r2[t + st]; }
        __syncthreads();
    }
    if (t == 0) {
        float inv = 1.f / (float)tot;
        float m = r1[0] * inv;
        float var = r2[0] * inv - m * m;
        stats[2 * o]     = m;
        stats[2 * o + 1] = rsqrtf(var + 1e-5f);
    }
}

// ---------------- 11) BN apply (in-place) + write first OUT channels to d_out ----------------
__global__ void bn_apply_kernel(float* __restrict__ buf, const float* __restrict__ stats,
                                const float* __restrict__ w, const float* __restrict__ bb,
                                float* __restrict__ outp, int N) {
    size_t idx = (size_t)blockIdx.x * blockDim.x + threadIdx.x;
    size_t tot = (size_t)B_ * CH * N;
    if (idx >= tot) return;
    int n = (int)(idx % N);
    int o = (int)((idx / N) % CH);
    int b = (int)(idx / ((size_t)N * CH));
    float v = (buf[idx] - stats[2 * o]) * stats[2 * o + 1] * w[o] + bb[o];
    buf[idx] = v;
    if (o < OUT_) outp[((size_t)b * OUT_ + o) * N + n] = v;
}

// ---------------- 12) I_XZ per batch (mean over S MC samples, sum over nodes) ----------------
__global__ void info_kernel(const float* __restrict__ buf, float* __restrict__ out,
                            int N, unsigned seed) {
    int b = blockIdx.x, n = threadIdx.x;
    float acc = 0.f;
    for (int c = 0; c < OUT_; ++c) {
        float mu  = buf[((size_t)b * CH + c) * N + n];
        float raw = buf[((size_t)b * CH + OUT_ + c) * N + n];
        float sp  = raw > 20.f ? raw : log1pf(__expf(raw));  // softplus
        float sd  = sp + 1e-10f;
        float ls  = __logf(sd);
        unsigned base = seed ^ (unsigned)(((b * N + n) * OUT_ + c) * 0x9E3779B1u);
        #pragma unroll
        for (int si = 0; si < S_; ++si) {
            float eps = hash_normal(base + (unsigned)si * 0x85EBCA6Bu);
            float Z = mu + sd * eps;
            acc += -0.5f * eps * eps - ls + 0.5f * Z * Z;     // logq - logp
        }
    }
    acc *= (1.f / (float)S_);
    __shared__ float red[256];
    red[n] = acc; __syncthreads();
    for (int st = N >> 1; st > 0; st >>= 1) {
        if (n < st) red[n] += red[n + st];
        __syncthreads();
    }
    if (n == 0) out[b] = red[0];
}

extern "C" void kernel_launch(void* const* d_in, const int* in_sizes, int n_in,
                              void* d_out, int out_size, void* d_ws, size_t ws_size,
                              hipStream_t stream) {
    const float* A_AP  = (const float*)d_in[0];
    const float* A_UE  = (const float*)d_in[1];
    const float* H     = (const float*)d_in[2];
    const float* Q1_AP = (const float*)d_in[3];
    const float* Q2_AP = (const float*)d_in[4];
    const float* Q1_UE = (const float*)d_in[5];
    const float* Q2_UE = (const float*)d_in[6];
    const float* P1_AP = (const float*)d_in[7];
    const float* P1_UE = (const float*)d_in[8];
    const float* AttAP = (const float*)d_in[9];
    const float* AttUE = (const float*)d_in[10];
    const float* bn_w  = (const float*)d_in[11];
    const float* bn_b  = (const float*)d_in[12];
    const int* G_AP    = (const int*)d_in[13];
    const int* G_UE    = (const int*)d_in[14];
    const int* GFAAP   = (const int*)d_in[15];
    const int* GFAUE   = (const int*)d_in[16];

    float* out        = (float*)d_out;
    float* out_ap     = out;
    float* out_ue     = out_ap + (size_t)B_ * OUT_ * NAP;
    float* out_ixz_ap = out_ue + (size_t)B_ * OUT_ * NUE;
    float* out_ixz_ue = out_ixz_ap + B_;
    float* out_iaz_ap = out_ixz_ue + B_;
    float* out_iaz_ue = out_iaz_ap + B_;

    char* w = (char*)d_ws;
    auto alloc = [&](size_t bytes) -> void* {
        void* p = (void*)w; w += (bytes + 255) & ~(size_t)255; return p;
    };
    float*     s_ap    = (float*)    alloc((size_t)B_ * NAP * 2 * 4);
    float*     s_ue    = (float*)    alloc((size_t)B_ * NUE * 2 * 4);
    float*     gate_ap = (float*)    alloc((size_t)B_ * EAP * 4);
    float*     alpha_ap= (float*)    alloc((size_t)B_ * EAP * 4);
    float*     gate_ue = (float*)    alloc((size_t)B_ * EUE * 4);
    float*     alpha_ue= (float*)    alloc((size_t)B_ * EUE * 4);
    _Float16*  AapT    = (_Float16*) alloc((size_t)B_ * NAP * D1 * 2);
    _Float16*  AueT    = (_Float16*) alloc((size_t)B_ * NUE * D1 * 2);
    _Float16*  HmApT   = (_Float16*) alloc((size_t)B_ * NAP * D2 * 2);
    _Float16*  HmUeT   = (_Float16*) alloc((size_t)B_ * NUE * D2 * 2);
    float*     AapMean = (float*)    alloc((size_t)B_ * D1 * 4);
    float*     AueMean = (float*)    alloc((size_t)B_ * D1 * 4);
    float*     bias_ap = (float*)    alloc((size_t)B_ * CH * 4);
    float*     bias_ue = (float*)    alloc((size_t)B_ * CH * 4);
    _Float16*  Qc_ap   = (_Float16*) alloc((size_t)CH * D1 * 2);
    _Float16*  Pc_ap   = (_Float16*) alloc((size_t)CH * D2 * 2);
    _Float16*  Qc_ue   = (_Float16*) alloc((size_t)CH * D1 * 2);
    _Float16*  Pc_ue   = (_Float16*) alloc((size_t)CH * D2 * 2);
    float*     apbuf   = (float*)    alloc((size_t)B_ * CH * NAP * 4);
    float*     uebuf   = (float*)    alloc((size_t)B_ * CH * NUE * 4);
    float*     stat_ap = (float*)    alloc((size_t)CH * 2 * 4);
    float*     stat_ue = (float*)    alloc((size_t)CH * 2 * 4);

    // 1) node scores
    node_score_kernel<<<(B_ * NAP + 255) / 256, 256, 0, stream>>>(A_AP, AttAP, s_ap, NAP);
    node_score_kernel<<<(B_ * NUE + 255) / 256, 256, 0, stream>>>(A_UE, AttUE, s_ue, NUE);
    // 2) edge gates
    edge_kernel<<<(B_ * EAP) / 256, 256, 0, stream>>>(s_ap, G_AP, gate_ap, alpha_ap, NAP, EAP, 0x1234567u);
    edge_kernel<<<(B_ * EUE) / 256, 256, 0, stream>>>(s_ue, G_UE, gate_ue, alpha_ue, NUE, EUE, 0x89abcdfu);
    // 3) I_AZ
    kl_kernel<<<B_, 256, 0, stream>>>(alpha_ap, out_iaz_ap, EAP);
    kl_kernel<<<B_, 256, 0, stream>>>(alpha_ue, out_iaz_ue, EUE);
    // 4) aggregation -> transposed f16
    agg_kernel<<<B_ * NAP, 64, 0, stream>>>(A_AP, gate_ap, G_AP, GFAAP, AapT, NAP, EAP);
    agg_kernel<<<B_ * NUE, 64, 0, stream>>>(A_UE, gate_ue, G_UE, GFAUE, AueT, NUE, EUE);
    // 5) feature means
    colmean_kernel<<<(B_ * D1 + 255) / 256, 256, 0, stream>>>(AapT, AapMean, NAP);
    colmean_kernel<<<(B_ * D1 + 255) / 256, 256, 0, stream>>>(AueT, AueMean, NUE);
    // 6) H means (the 268 MB streaming pass)
    hmean_kernel<<<B_ * D2, 256, 0, stream>>>(H, HmApT, HmUeT);
    // 7) weight conversion (scale folded)
    convw_kernel<<<(CH * D1 + 255) / 256, 256, 0, stream>>>(Q1_AP, P1_AP, Qc_ap, Pc_ap);
    convw_kernel<<<(CH * D1 + 255) / 256, 256, 0, stream>>>(Q1_UE, P1_UE, Qc_ue, Pc_ue);
    // 8) cross biases (AP uses UE mean and vice versa)
    bias_kernel<<<(B_ * CH + 255) / 256, 256, 0, stream>>>(Q2_AP, AueMean, bias_ap);
    bias_kernel<<<(B_ * CH + 255) / 256, 256, 0, stream>>>(Q2_UE, AapMean, bias_ue);
    // 9) WMMA GEMMs + ReLU (4 N-tiles per wave, 12 interleaved wmma)
    gemm_wmma_kernel<NAP><<<(B_ * (CH / 16) * (NAP / 64)) / 8, 256, 0, stream>>>(
        Qc_ap, Pc_ap, AapT, HmApT, bias_ap, apbuf);
    gemm_wmma_kernel<NUE><<<(B_ * (CH / 16) * (NUE / 64)) / 8, 256, 0, stream>>>(
        Qc_ue, Pc_ue, AueT, HmUeT, bias_ue, uebuf);
    // 10) BN stats
    bn_stats_kernel<<<CH, 256, 0, stream>>>(apbuf, stat_ap, NAP);
    bn_stats_kernel<<<CH, 256, 0, stream>>>(uebuf, stat_ue, NUE);
    // 11) BN apply + main outputs
    bn_apply_kernel<<<(int)(((size_t)B_ * CH * NAP + 255) / 256), 256, 0, stream>>>(
        apbuf, stat_ap, bn_w, bn_b, out_ap, NAP);
    bn_apply_kernel<<<(int)(((size_t)B_ * CH * NUE + 255) / 256), 256, 0, stream>>>(
        uebuf, stat_ue, bn_w, bn_b, out_ue, NUE);
    // 12) I_XZ
    info_kernel<<<B_, NAP, 0, stream>>>(apbuf, out_ixz_ap, NAP, 0xbeefcafeu);
    info_kernel<<<B_, NUE, 0, stream>>>(uebuf, out_ixz_ue, NUE, 0xfacefeedu);
}
